// TGMamba_14809047236987
// MI455X (gfx1250) — matrix operand order repaired and checked by reading, not compiled, not run on hardware
//
#include <hip/hip_runtime.h>
#include <hip/hip_fp16.h>

typedef __attribute__((ext_vector_type(16))) _Float16 v16h;
typedef __attribute__((ext_vector_type(8)))  float    v8f;

#define LSEQ  1024
#define DM    16
#define DI    32
#define DS    16
#define DCONV 4
#define CH    8     // scan chunk (timesteps staged per LDS buffer)

__device__ __forceinline__ float silu_f(float v) { return v / (1.f + __expf(-v)); }

// ---------- WMMA fragment helpers (CDNA5 16x16x32 f16 layouts) ----------
// A (16xK, K<=32) f16: lane L holds row M=L%16; hi=L/16.
//   element e -> K = 16*(e>>3) + (e&7) + 8*hi   (per ISA 16-bit A 16x32 table)
__device__ __forceinline__ v16h a_frag_lds(const float* s, int ld, int kvalid) {
    int lane = threadIdx.x & 31;
    int row = lane & 15, hi = lane >> 4;
    v16h a;
#pragma unroll
    for (int e = 0; e < 16; ++e) {
        int k = ((e >> 3) << 4) + (e & 7) + (hi << 3);
        a[e] = (k < kvalid) ? (_Float16)s[row * ld + k] : (_Float16)0.f;
    }
    return a;
}

// B (KxN, N=16) built from row-major weight W[nout][K]: B[k][n] = W[r0+n][k].
// lane L holds column N=L%16; element e -> K = 16*(e>>3) + (e&7) + 8*(L/16).
__device__ __forceinline__ v16h b_frag_w(const float* W, int ldk, int r0, int kvalid) {
    int lane = threadIdx.x & 31;
    int col = lane & 15, hi = lane >> 4;
    v16h b;
#pragma unroll
    for (int e = 0; e < 16; ++e) {
        int k = ((e >> 3) << 4) + (e & 7) + (hi << 3);
        b[e] = (k < kvalid) ? (_Float16)W[(r0 + col) * ldk + k] : (_Float16)0.f;
    }
    return b;
}

__device__ __forceinline__ v8f wmma_f16(v16h a, v16h b, v8f c) {
    return __builtin_amdgcn_wmma_f32_16x16x32_f16(false, a, false, b, (short)0, c,
                                                  false, false);
}

// ---------- GCN normalization prep ----------
__global__ void k_deg_init(float* deg, int N) {
    int i = blockIdx.x * blockDim.x + threadIdx.x;
    if (i < N) deg[i] = 1.f;  // self-loop weight
}
__global__ void k_deg_acc(const int* __restrict__ ei, const float* __restrict__ w,
                          int E, float* deg) {
    int e = blockIdx.x * blockDim.x + threadIdx.x;
    if (e < E) atomicAdd(&deg[ei[E + e]], w[e]);
}
__global__ void k_dinv(float* deg, int N) {
    int i = blockIdx.x * blockDim.x + threadIdx.x;
    if (i < N) { float v = deg[i]; deg[i] = (v > 0.f) ? rsqrtf(v) : 0.f; }
}
__global__ void k_coef(const int* __restrict__ ei, const float* __restrict__ w,
                       const float* __restrict__ dinv, int E, int N,
                       float* cE, float* cS) {
    int i = blockIdx.x * blockDim.x + threadIdx.x;
    if (i < E) {
        cE[i] = dinv[ei[i]] * w[i] * dinv[ei[E + i]];
    } else if (i < E + N) {
        float d = dinv[i - E];
        cS[i - E] = d * d;
    }
}

// ---------- Front: in_proj (WMMA) -> causal conv+SiLU -> x_proj (WMMA) -> delta ----------
__global__ __launch_bounds__(128) void k_front(
    const float* __restrict__ hidden, const float* __restrict__ in_proj_w,
    const float* __restrict__ conv_w, const float* __restrict__ conv_b,
    const float* __restrict__ x_proj_w, const float* __restrict__ dt_proj_w,
    const float* __restrict__ dt_proj_b,
    float* __restrict__ xbuf, float* __restrict__ zbuf, float* __restrict__ dbuf,
    float* __restrict__ Braw, float* __restrict__ Craw) {
    __shared__ float sHid[19 * DM];   // rows 0..2: halo t0-3..t0-1; rows 3..18: tile
    __shared__ float sXZ[16 * 64];    // raw in_proj output (x | z)
    __shared__ float sHaloX[3 * DI];  // raw x for conv halo
    __shared__ float sX[16 * DI];     // conv+SiLU'd x
    __shared__ float sDt[16];

    int n = blockIdx.x >> 6;
    int t0 = (blockIdx.x & 63) << 4;
    int tid = threadIdx.x;
    int wave = tid >> 5;

    // Preload weight fragments (per-wave output tile)
    v16h bIn = b_frag_w(in_proj_w, DM, 16 * wave, DM);         // (64,16): K=16
    v16h bXp = {};
    if (wave < 2) bXp = b_frag_w(x_proj_w, DI, 1 + 16 * wave, DI);  // rows 1..32, K=32

    // Load hidden rows t0-3 .. t0+15
    for (int p = tid; p < 19 * DM; p += 128) {
        int r = p >> 4, c = p & 15;
        int tg = t0 + r - 3;
        sHid[p] = (tg >= 0) ? hidden[((size_t)n * LSEQ + tg) * DM + c] : 0.f;
    }
    __syncthreads();

    // GEMM1: 16 tokens x 16 -> 64 (x|z); each wave does one 16-col tile
    {
        v16h a = a_frag_lds(sHid + 3 * DM, DM, DM);
        v8f c = {};
        c = wmma_f16(a, bIn, c);
        int lane = tid & 31, col = lane & 15, hi = lane >> 4;
#pragma unroll
        for (int v = 0; v < 8; ++v)
            sXZ[(v + 8 * hi) * 64 + 16 * wave + col] = c[v];
    }
    // Halo raw-x (3 rows x 32 chan) via VALU dot-16 (makes tiles independent)
    if (tid < 96) {
        int r = tid >> 5, ch = tid & 31;
        float acc = 0.f;
#pragma unroll
        for (int k = 0; k < DM; ++k) acc += sHid[r * DM + k] * in_proj_w[ch * DM + k];
        sHaloX[r * DI + ch] = acc;
    }
    __syncthreads();

    // Depthwise causal conv + SiLU; emit x and z
    for (int p = tid; p < 16 * DI; p += 128) {
        int t = p >> 5, ch = p & 31;
        float acc = conv_b[ch];
#pragma unroll
        for (int k = 0; k < DCONV; ++k) {
            int tl = t + k - (DCONV - 1);
            float xv;
            if (tl >= 0) xv = sXZ[tl * 64 + ch];
            else xv = (t0 + tl >= 0) ? sHaloX[(tl + 3) * DI + ch] : 0.f;
            acc += xv * conv_w[ch * DCONV + k];
        }
        float s = silu_f(acc);
        sX[t * DI + ch] = s;
        size_t g = ((size_t)n * LSEQ + t0 + t) * DI + ch;
        xbuf[g] = s;
        zbuf[g] = sXZ[t * 64 + DI + ch];
    }
    __syncthreads();

    // GEMM2 (waves 0/1: Braw/Craw) + dt row (wave 2, lanes 0..15)
    if (wave < 2) {
        v16h a = a_frag_lds(sX, DI, DI);
        v8f c = {};
        c = wmma_f16(a, bXp, c);
        float* dstb = wave ? Craw : Braw;
        int lane = tid & 31, col = lane & 15, hi = lane >> 4;
#pragma unroll
        for (int v = 0; v < 8; ++v)
            dstb[((size_t)n * LSEQ + t0 + v + 8 * hi) * DS + col] = c[v];
    } else if (tid < 64 + 16) {
        int t = tid - 64;
        float acc = 0.f;
#pragma unroll
        for (int ch = 0; ch < DI; ++ch) acc += sX[t * DI + ch] * x_proj_w[ch];  // row 0
        sDt[t] = acc;
    }
    __syncthreads();

    // delta = softplus(dt * dt_proj_w + dt_proj_b)
    for (int p = tid; p < 16 * DI; p += 128) {
        int t = p >> 5, d = p & 31;
        float v = sDt[t] * dt_proj_w[d] + dt_proj_b[d];
        float sp = (v > 20.f) ? v : log1pf(__expf(v));
        dbuf[((size_t)n * LSEQ + t0 + t) * DI + d] = sp;
    }
}

// ---------- GCN scatter (atomics; L2-resident operands) ----------
__global__ void k_zero(float* p, size_t nel) {
    size_t i = (size_t)blockIdx.x * blockDim.x + threadIdx.x;
    size_t st = (size_t)gridDim.x * blockDim.x;
    for (; i < nel; i += st) p[i] = 0.f;
}

__global__ __launch_bounds__(256) void k_scatter(
    const int* __restrict__ ei, const float* __restrict__ coefE,
    const float* __restrict__ coefS, int E,
    const float* __restrict__ Braw, const float* __restrict__ Craw,
    const float* __restrict__ xbuf,
    float* __restrict__ aggB, float* __restrict__ aggC, float* __restrict__ aggU) {
    int e = blockIdx.x;
    int s, d;
    float cf;
    if (e < E) { s = ei[e]; d = ei[E + e]; cf = coefE[e]; }
    else       { s = d = e - E;            cf = coefS[s]; }
    size_t sb16 = (size_t)s * LSEQ * DS, db16 = (size_t)d * LSEQ * DS;
    size_t sb32 = (size_t)s * LSEQ * DI, db32 = (size_t)d * LSEQ * DI;

    // Pull source rows toward the WGP ahead of the atomic stream
    // (global_prefetch_b8; no counters, no VGPR return).
    for (int off = threadIdx.x * 16; off < LSEQ * DS; off += 256 * 16) {
        __builtin_prefetch(&Braw[sb16 + off], 0, 0);
        __builtin_prefetch(&Craw[sb16 + off], 0, 0);
    }
    for (int off = threadIdx.x * 16; off < LSEQ * DI; off += 256 * 16) {
        __builtin_prefetch(&xbuf[sb32 + off], 0, 0);
    }

    for (int p = threadIdx.x; p < LSEQ * 64; p += 256) {
        int t = p >> 6, f = p & 63;
        if (f < 16)
            atomicAdd(&aggB[db16 + t * DS + f], Braw[sb16 + t * DS + f] * cf);
        else if (f < 32)
            atomicAdd(&aggC[db16 + t * DS + (f - 16)], Craw[sb16 + t * DS + (f - 16)] * cf);
        else
            atomicAdd(&aggU[db32 + t * DI + (f - 32)], xbuf[sb32 + t * DI + (f - 32)] * cf);
    }
}

// ---------- GCN apply: agg @ W^T then SiLU (in-place per 16-row tile, WMMA) ----------
__global__ __launch_bounds__(128) void k_gcn_apply(
    const float* __restrict__ W_A, const float* __restrict__ W_B,
    const float* __restrict__ W_C,
    float* __restrict__ aggB, float* __restrict__ aggC, float* __restrict__ aggU) {
    __shared__ float sB[16 * DS], sC[16 * DS], sU[16 * DI];
    int n = blockIdx.x >> 6;
    int t0 = (blockIdx.x & 63) << 4;
    int tid = threadIdx.x, wave = tid >> 5;
    size_t b16 = ((size_t)n * LSEQ + t0) * DS;
    size_t b32 = ((size_t)n * LSEQ + t0) * DI;
    for (int p = tid; p < 16 * DS; p += 128) { sB[p] = aggB[b16 + p]; sC[p] = aggC[b16 + p]; }
    for (int p = tid; p < 16 * DI; p += 128) { sU[p] = aggU[b32 + p]; }
    __syncthreads();
    int lane = tid & 31, col = lane & 15, hi = lane >> 4;
    v8f c = {};
    if (wave == 0) {
        v16h a = a_frag_lds(sB, DS, DS);
        v16h b = b_frag_w(W_A, DS, 0, DS);
        c = wmma_f16(a, b, c);
#pragma unroll
        for (int v = 0; v < 8; ++v) aggB[b16 + (v + 8 * hi) * DS + col] = silu_f(c[v]);
    } else if (wave == 1) {
        v16h a = a_frag_lds(sC, DS, DS);
        v16h b = b_frag_w(W_C, DS, 0, DS);
        c = wmma_f16(a, b, c);
#pragma unroll
        for (int v = 0; v < 8; ++v) aggC[b16 + (v + 8 * hi) * DS + col] = silu_f(c[v]);
    } else {
        int nt = wave - 2;  // output cols 16*nt .. 16*nt+15
        v16h a = a_frag_lds(sU, DI, DI);
        v16h b = b_frag_w(W_B, DI, 16 * nt, DI);
        c = wmma_f16(a, b, c);
#pragma unroll
        for (int v = 0; v < 8; ++v)
            aggU[b32 + (v + 8 * hi) * DI + 16 * nt + col] = silu_f(c[v]);
    }
}

// ---------- Selective scan fused with D-skip, gating, RMSNorm, out_proj ----------
// One wave32 per node; lane d owns channel d with 16-state recurrence in registers.
// Software-pipelined: next chunk loaded to registers (global_load_b128) while the
// current chunk computes from a double-buffered LDS set; out_proj hoisted to
// once-per-chunk (2 barriers / 8 timesteps instead of 2 / timestep).
struct ChunkRegs {
    float4 dv0, dv1, uv0, uv1, xv0, xv1, zv0, zv1, bv, cv;
};

__device__ __forceinline__ void load_chunk(
    const float* __restrict__ dbuf, const float* __restrict__ ug,
    const float* __restrict__ xbuf, const float* __restrict__ zbuf,
    const float* __restrict__ Bg, const float* __restrict__ Cg,
    size_t base32, size_t base16, int tc, int d, ChunkRegs& r) {
    const float4* pD = (const float4*)(dbuf + base32 + (size_t)tc * DI);
    const float4* pU = (const float4*)(ug   + base32 + (size_t)tc * DI);
    const float4* pX = (const float4*)(xbuf + base32 + (size_t)tc * DI);
    const float4* pZ = (const float4*)(zbuf + base32 + (size_t)tc * DI);
    r.dv0 = pD[d]; r.dv1 = pD[d + 32];
    r.uv0 = pU[d]; r.uv1 = pU[d + 32];
    r.xv0 = pX[d]; r.xv1 = pX[d + 32];
    r.zv0 = pZ[d]; r.zv1 = pZ[d + 32];
    r.bv = ((const float4*)(Bg + base16 + (size_t)tc * DS))[d];
    r.cv = ((const float4*)(Cg + base16 + (size_t)tc * DS))[d];
}

__device__ __forceinline__ void stage_chunk(
    float* sD, float* sU, float* sX, float* sZ, float* sB, float* sC,
    int d, const ChunkRegs& r) {
    ((float4*)sD)[d] = r.dv0; ((float4*)sD)[d + 32] = r.dv1;
    ((float4*)sU)[d] = r.uv0; ((float4*)sU)[d + 32] = r.uv1;
    ((float4*)sX)[d] = r.xv0; ((float4*)sX)[d + 32] = r.xv1;
    ((float4*)sZ)[d] = r.zv0; ((float4*)sZ)[d + 32] = r.zv1;
    ((float4*)sB)[d] = r.bv;  ((float4*)sC)[d] = r.cv;
}

__global__ __launch_bounds__(32) void k_scan(
    const float* __restrict__ dbuf, const float* __restrict__ Bg,
    const float* __restrict__ Cg, const float* __restrict__ ug,
    const float* __restrict__ xbuf, const float* __restrict__ zbuf,
    const float* __restrict__ A_log, const float* __restrict__ D_skip,
    const float* __restrict__ norm_w, const float* __restrict__ out_proj_w,
    float* __restrict__ out) {
    __shared__ __align__(16) float sD[2][CH * DI], sU[2][CH * DI];
    __shared__ __align__(16) float sX[2][CH * DI], sZ[2][CH * DI];
    __shared__ __align__(16) float sB[2][CH * DS], sC[2][CH * DS];
    __shared__ float sG[CH * DI];
    __shared__ float sW[DM * DI];
    int n = blockIdx.x;
    int d = threadIdx.x;  // 0..31 == channel
    for (int p = d; p < DM * DI; p += 32) sW[p] = out_proj_w[p];
    float Ar[DS], h[DS];
#pragma unroll
    for (int s = 0; s < DS; ++s) { Ar[s] = -__expf(A_log[d * DS + s]); h[s] = 0.f; }
    float dsk = D_skip[d], nw = norm_w[d];
    size_t base32 = (size_t)n * LSEQ * DI;
    size_t base16 = (size_t)n * LSEQ * DS;

    ChunkRegs r;
    load_chunk(dbuf, ug, xbuf, zbuf, Bg, Cg, base32, base16, 0, d, r);
    stage_chunk(sD[0], sU[0], sX[0], sZ[0], sB[0], sC[0], d, r);
    __syncthreads();

    int cur = 0;
    for (int tc = 0; tc < LSEQ; tc += CH) {
        bool more = (tc + CH) < LSEQ;
        if (more)  // issue next chunk's global loads; consumed after compute
            load_chunk(dbuf, ug, xbuf, zbuf, Bg, Cg, base32, base16, tc + CH, d, r);

        const float* cD = sD[cur]; const float* cU = sU[cur];
        const float* cX = sX[cur]; const float* cZ = sZ[cur];
        const float* cB = sB[cur]; const float* cC = sC[cur];
#pragma unroll 1
        for (int tt = 0; tt < CH; ++tt) {
            float dlt = cD[tt * DI + d];
            float du = dlt * cU[tt * DI + d];
            float y = 0.f;
#pragma unroll
            for (int s = 0; s < DS; ++s) {
                h[s] = __expf(dlt * Ar[s]) * h[s] + du * cB[tt * DS + s];
                y += h[s] * cC[tt * DS + s];
            }
            float yd = y + cX[tt * DI + d] * dsk;
            float g = yd * silu_f(cZ[tt * DI + d]);
            float ss = g * g;
#pragma unroll
            for (int m = 16; m >= 1; m >>= 1) ss += __shfl_xor(ss, m, 32);
            float rn = rsqrtf(ss * (1.f / DI) + 1e-5f);
            sG[tt * DI + d] = g * rn * nw;
        }
        __syncthreads();

        // out_proj for the whole chunk: CH*16 outputs over 32 lanes (4 dot-32 each)
        for (int p = d; p < CH * DM; p += 32) {
            int tt = p >> 4, j = p & 15;
            float o = 0.f;
#pragma unroll
            for (int c = 0; c < DI; ++c) o += sG[tt * DI + c] * sW[j * DI + c];
            out[((size_t)n * LSEQ + tc + tt) * DM + j] = o;
        }
        if (more) {
            cur ^= 1;
            stage_chunk(sD[cur], sU[cur], sX[cur], sZ[cur], sB[cur], sC[cur], d, r);
        }
        __syncthreads();
    }
}

extern "C" void kernel_launch(void* const* d_in, const int* in_sizes, int n_in,
                              void* d_out, int out_size, void* d_ws, size_t ws_size,
                              hipStream_t stream) {
    (void)n_in; (void)out_size; (void)ws_size;
    const float* hidden     = (const float*)d_in[0];
    const int*   ei         = (const int*)d_in[1];
    const float* ew         = (const float*)d_in[2];
    const float* in_proj_w  = (const float*)d_in[3];
    const float* conv_w     = (const float*)d_in[4];
    const float* conv_b     = (const float*)d_in[5];
    const float* x_proj_w   = (const float*)d_in[6];
    const float* dt_proj_w  = (const float*)d_in[7];
    const float* dt_proj_b  = (const float*)d_in[8];
    const float* A_log      = (const float*)d_in[9];
    const float* D_skip     = (const float*)d_in[10];
    const float* W_A        = (const float*)d_in[11];
    const float* W_B        = (const float*)d_in[12];
    const float* W_C        = (const float*)d_in[13];
    const float* norm_w     = (const float*)d_in[14];
    const float* out_proj_w = (const float*)d_in[15];
    float* out = (float*)d_out;

    int N = in_sizes[0] / (LSEQ * DM);
    int E = in_sizes[1] / 2;

    float* ws = (float*)d_ws;
    size_t off = 0;
    float* deg   = ws + off; off += (size_t)N;
    float* coefE = ws + off; off += (size_t)E;
    float* coefS = ws + off; off += (size_t)N;
    off = (off + 255) & ~(size_t)255;
    size_t n32 = (size_t)N * LSEQ * DI;
    size_t n16 = (size_t)N * LSEQ * DS;
    float* zbuf = ws + off; off += n32;
    float* xbuf = ws + off; off += n32;
    float* dbuf = ws + off; off += n32;
    float* Braw = ws + off; off += n16;
    float* Craw = ws + off; off += n16;
    float* aggB = ws + off; off += n16;  // aggB/aggC/aggU contiguous
    float* aggC = ws + off; off += n16;
    float* aggU = ws + off; off += n32;

    // GCN normalization coefficients
    k_deg_init<<<(N + 255) / 256, 256, 0, stream>>>(deg, N);
    k_deg_acc<<<(E + 255) / 256, 256, 0, stream>>>(ei, ew, E, deg);
    k_dinv<<<(N + 255) / 256, 256, 0, stream>>>(deg, N);
    k_coef<<<(E + N + 255) / 256, 256, 0, stream>>>(ei, ew, deg, E, N, coefE, coefS);

    // Fused front (in_proj + conv/SiLU + x_proj + delta), WMMA per tile
    k_front<<<N * 64, 128, 0, stream>>>(hidden, in_proj_w, conv_w, conv_b, x_proj_w,
                                        dt_proj_w, dt_proj_b, xbuf, zbuf, dbuf,
                                        Braw, Craw);

    // GCN message passing
    k_zero<<<2048, 256, 0, stream>>>(aggB, n16 + n16 + n32);
    k_scatter<<<E + N, 256, 0, stream>>>(ei, coefE, coefS, E, Braw, Craw, xbuf,
                                         aggB, aggC, aggU);
    k_gcn_apply<<<N * 64, 128, 0, stream>>>(W_A, W_B, W_C, aggB, aggC, aggU);

    // Selective scan fused with gating + RMSNorm + out_proj (software-pipelined)
    k_scan<<<N, 32, 0, stream>>>(dbuf, aggB, aggC, aggU, xbuf, zbuf, A_log, D_skip,
                                 norm_w, out_proj_w, out);
}
